// NSF_CL_62929860821556
// MI455X (gfx1250) — matrix operand order repaired and checked by reading, not compiled, not run on hardware
//
#include <hip/hip_runtime.h>
#include <hip/hip_bf16.h>

typedef __attribute__((ext_vector_type(16))) __bf16 v16bf;
typedef __attribute__((ext_vector_type(8)))  __bf16 v8bf;
typedef __attribute__((ext_vector_type(8)))  float  v8f;

#define ROWS      32          // rows per workgroup
#define NH        512
#define NOUT      736
#define SA        (NH + 8)    // LDS activation stride (bf16 elems), 16B padded
#define SA1       (32 + 8)    // layer-1 input stride
#define SOUT      (NOUT + 8)  // f32 staging stride
#define LDS_A_OFF 0
#define LDS_B_OFF (ROWS * SA * 2)                 // 33280
#define LDS_O_OFF (LDS_B_OFF * 2)                 // 66560
#define LDS_BYTES (LDS_O_OFF + ROWS * SOUT * 4)   // 161792

// packed bf16 weight offsets (elements) within one net's region
#define OFF_L1 0
#define OFF_L2 16384
#define OFF_L3 278528
#define OFF_L4 540672
#define NET_ELEMS 917504
#define LD1_WS_BYTE_OFF (2 * NET_ELEMS * 2)       // 3,670,016 (16B aligned)

__device__ __forceinline__ float softplusf(float x) {
    return x > 15.0f ? x : log1pf(__expf(x));
}

// 2*BND*softmax(logits) -> softmax again -> affine -> cumsum knots in [-3,3]
__device__ __forceinline__ void knots8(const float* p, float* c /*9*/, float* w /*8*/) {
    float t[8];
    float mx = p[0];
#pragma unroll
    for (int i = 1; i < 8; ++i) mx = fmaxf(mx, p[i]);
    float s = 0.f;
#pragma unroll
    for (int i = 0; i < 8; ++i) { t[i] = __expf(p[i] - mx); s += t[i]; }
    float inv = 1.f / s;
    float mx2 = -1e30f;
#pragma unroll
    for (int i = 0; i < 8; ++i) { t[i] = 6.f * t[i] * inv; mx2 = fmaxf(mx2, t[i]); }
    s = 0.f;
#pragma unroll
    for (int i = 0; i < 8; ++i) { t[i] = __expf(t[i] - mx2); s += t[i]; }
    inv = 1.f / s;
    float run = 0.f;
    c[0] = -3.f;
#pragma unroll
    for (int i = 0; i < 8; ++i) {
        float wi = 0.001f + (1.f - 0.001f * 8.f) * t[i] * inv;
        run += wi;
        c[i + 1] = 6.f * run - 3.f;
    }
    c[8] = 3.f;
#pragma unroll
    for (int i = 0; i < 8; ++i) w[i] = c[i + 1] - c[i];
}

// fp32 [K][N] -> bf16 packed per-(16x32) WMMA B-tile: lane-major 32B chunks
__global__ void nsf_pack_w(const float* __restrict__ w, __bf16* __restrict__ dst,
                           int N, int K) {
    int e = blockIdx.x * 256 + threadIdx.x;
    if (e >= N * K) return;
    int KB = K >> 5;
    int blk = e >> 9;          // 512 bf16 per block
    int within = e & 511;
    int lane = within >> 4;
    int j = within & 15;
    int nt = blk / KB, kb = blk % KB;
    int n = nt * 16 + (lane & 15);
    int k = kb * 32 + ((j >> 3) << 4) + ((lane >> 4) << 3) + (j & 7);
    dst[e] = (__bf16)w[(size_t)k * N + n];
}

__device__ __forceinline__ v16bf ldfragA(const __bf16* p) {
    v8bf lo = *(const v8bf*)(p);
    v8bf hi = *(const v8bf*)(p + 16);
    return __builtin_shufflevector(lo, hi, 0,1,2,3,4,5,6,7,8,9,10,11,12,13,14,15);
}
__device__ __forceinline__ v16bf ldfragB(const __bf16* p) {
    v8bf lo = *(const v8bf*)(p);
    v8bf hi = *(const v8bf*)(p + 8);
    return __builtin_shufflevector(lo, hi, 0,1,2,3,4,5,6,7,8,9,10,11,12,13,14,15);
}

// One GEMM layer on a 32-row tile. 256 threads = 8 waves.
// 2x2 register tiling: per wave iteration, 2 A-frags x 2 B-frags -> 4 WMMAs.
__device__ __forceinline__ void gemm_layer(
    const __bf16* __restrict__ a_lds, int sa,
    const __bf16* __restrict__ wpk, const float* __restrict__ bias,
    int N, int K, __bf16* o_lds, int so, bool lrelu,
    float* o_f32, int sof) {
    const int lane = threadIdx.x & 31;
    const int wave = threadIdx.x >> 5;
    const int KB = K >> 5;
    const int NP = N >> 5;      // pairs of 16-wide n-tiles (N is a multiple of 32)
    const int ml = lane & 15;
    const int g  = lane >> 4;

    for (int p = wave; p < NP; p += 8) {
        const int nt0 = 2 * p;
        v8f acc00 = {0.f,0.f,0.f,0.f,0.f,0.f,0.f,0.f};  // mt0 x nt0
        v8f acc10 = acc00;                              // mt1 x nt0
        v8f acc01 = acc00;                              // mt0 x nt1
        v8f acc11 = acc00;                              // mt1 x nt1
        const __bf16* b0 = wpk + (((size_t)nt0 * KB) << 9) + (lane << 4);
        const __bf16* b1 = b0 + ((size_t)KB << 9);
        const __bf16* a0 = a_lds + ml * sa + g * 8;
        const __bf16* a1 = a0 + 16 * sa;
#pragma unroll 2
        for (int kb = 0; kb < KB; ++kb) {
            v16bf af0 = ldfragA(a0);
            v16bf af1 = ldfragA(a1);
            v16bf bf0 = ldfragB(b0);
            v16bf bf1 = ldfragB(b1);
            acc00 = __builtin_amdgcn_wmma_f32_16x16x32_bf16(false, af0, false, bf0, (short)0, acc00, false, false);
            acc10 = __builtin_amdgcn_wmma_f32_16x16x32_bf16(false, af1, false, bf0, (short)0, acc10, false, false);
            acc01 = __builtin_amdgcn_wmma_f32_16x16x32_bf16(false, af0, false, bf1, (short)0, acc01, false, false);
            acc11 = __builtin_amdgcn_wmma_f32_16x16x32_bf16(false, af1, false, bf1, (short)0, acc11, false, false);
            a0 += 32; a1 += 32; b0 += 512; b1 += 512;
        }
        // store 4 tiles
        auto store_tile = [&](const v8f& acc, int mt, int nt) {
            int n = nt * 16 + ml;
            float bn = bias[n];
            if (o_f32) {
                float* op = o_f32 + n;
#pragma unroll
                for (int r = 0; r < 8; ++r) {
                    int m = mt * 16 + g * 8 + r;
                    op[m * sof] = acc[r] + bn;
                }
            } else {
                __bf16* op = o_lds + n;
#pragma unroll
                for (int r = 0; r < 8; ++r) {
                    int m = mt * 16 + g * 8 + r;
                    float v = acc[r] + bn;
                    if (lrelu) v = v > 0.f ? v : 0.2f * v;
                    op[m * so] = (__bf16)v;
                }
            }
        };
        store_tile(acc00, 0, nt0);
        store_tile(acc10, 1, nt0);
        store_tile(acc01, 0, nt0 + 1);
        store_tile(acc11, 1, nt0 + 1);
    }
}

// Fused: bf16-WMMA 4-layer MLP on mlp_in columns + RQS coupling of y_in columns.
__global__ void __launch_bounds__(256)
nsf_half_kernel(const float* __restrict__ mlp_in, int mlp_coloff,
                const float* __restrict__ y_in, int y_coloff,
                const __bf16* __restrict__ wpk,
                const float* __restrict__ b1, const float* __restrict__ b2,
                const float* __restrict__ b3, const float* __restrict__ b4,
                float* __restrict__ out_y, int out_coloff,
                float* __restrict__ ld_ws, float* __restrict__ logdet_out) {
    extern __shared__ char smem[];
    __bf16* bufA = (__bf16*)(smem + LDS_A_OFF);
    __bf16* bufB = (__bf16*)(smem + LDS_B_OFF);
    float*  outf = (float*)(smem + LDS_O_OFF);

    const int row0 = blockIdx.x * ROWS;

    // stage 0: load 32x32 fp32 input, convert to bf16 (stride SA1)
    for (int e = threadIdx.x; e < ROWS * 32; e += 256) {
        int r = e >> 5, c = e & 31;
        float v = mlp_in[(size_t)(row0 + r) * 64 + mlp_coloff + c];
        bufA[r * SA1 + c] = (__bf16)v;
    }
    __syncthreads();

    gemm_layer(bufA, SA1, wpk + OFF_L1, b1, NH, 32,  bufB, SA, true,  nullptr, 0);
    __syncthreads();
    gemm_layer(bufB, SA,  wpk + OFF_L2, b2, NH, NH,  bufA, SA, true,  nullptr, 0);
    __syncthreads();
    gemm_layer(bufA, SA,  wpk + OFF_L3, b3, NH, NH,  bufB, SA, true,  nullptr, 0);
    __syncthreads();
    gemm_layer(bufB, SA,  wpk + OFF_L4, b4, NOUT, NH, nullptr, 0, false, outf, SOUT);
    __syncthreads();

    // RQS spline epilogue: 8 threads per row, 4 dims per thread
    const int t   = threadIdx.x;
    const int row = t >> 3;
    const int l8  = t & 7;
    const size_t nrow = (size_t)row0 + row;
    const float* Prow = outf + row * SOUT;
    const float* yrow = y_in + nrow * 64 + y_coloff;
    float* orow = out_y + nrow * 64 + out_coloff;
    float ldsum = 0.f;
#pragma unroll
    for (int ii = 0; ii < 4; ++ii) {
        int d = l8 + ii * 8;
        const float* P = Prow + d * 23;
        float cw[9], wid[8], ch[9], hei[8], der[9];
        knots8(P, cw, wid);
        knots8(P + 8, ch, hei);
        der[0] = 1.0f; der[8] = 1.0f;
#pragma unroll
        for (int i = 0; i < 7; ++i) {
            float d1 = softplusf(P[16 + i]);
            der[i + 1] = 0.001f + softplusf(d1);
        }
        float x = yrow[d];
        bool inside = (x >= -3.f) && (x <= 3.f);
        float xc = fminf(fmaxf(x, -3.f), 3.f);
        int idx = -1;
#pragma unroll
        for (int i = 0; i < 9; ++i) {
            float edge = cw[i] + (i == 8 ? 1e-6f : 0.f);
            idx += (xc >= edge) ? 1 : 0;
        }
        idx = idx < 0 ? 0 : (idx > 7 ? 7 : idx);
        float in_cw = cw[idx], in_w = wid[idx], in_ch = ch[idx], in_h = hei[idx];
        float dlt = in_h / in_w;
        float dd = der[idx], dd1 = der[idx + 1];
        float th = (xc - in_cw) / in_w;
        float t1 = th * (1.f - th);
        float num = in_h * (dlt * th * th + dd * t1);
        float den = dlt + (dd + dd1 - 2.f * dlt) * t1;
        float outv = in_ch + num / den;
        float om = 1.f - th;
        float dnum = dlt * dlt * (dd1 * th * th + 2.f * dlt * t1 + dd * om * om);
        float lad = __logf(dnum) - 2.f * __logf(den);
        orow[d] = inside ? outv : x;
        ldsum += inside ? lad : 0.f;
    }
    ldsum += __shfl_xor(ldsum, 1);
    ldsum += __shfl_xor(ldsum, 2);
    ldsum += __shfl_xor(ldsum, 4);
    if (l8 == 0) {
        if (logdet_out) logdet_out[nrow] = ld_ws[nrow] + ldsum;
        else            ld_ws[nrow] = ldsum;
    }
}

extern "C" void kernel_launch(void* const* d_in, const int* in_sizes, int n_in,
                              void* d_out, int out_size, void* d_ws, size_t ws_size,
                              hipStream_t stream) {
    const float* x = (const float*)d_in[0];
    const int nrows = in_sizes[0] / 64;

    __bf16* wp1 = (__bf16*)d_ws;
    __bf16* wp2 = wp1 + NET_ELEMS;
    float* ld1  = (float*)((char*)d_ws + LD1_WS_BYTE_OFF);
    float* yout = (float*)d_out;
    float* logdet = yout + (size_t)nrows * 64;

    // pack weights fp32 [in][out] -> bf16 WMMA-fragment order
    struct { int src; __bf16* dst; int N; int K; } packs[8] = {
        {1,  wp1 + OFF_L1, NH,   32}, {3,  wp1 + OFF_L2, NH,   NH},
        {5,  wp1 + OFF_L3, NH,   NH}, {7,  wp1 + OFF_L4, NOUT, NH},
        {9,  wp2 + OFF_L1, NH,   32}, {11, wp2 + OFF_L2, NH,   NH},
        {13, wp2 + OFF_L3, NH,   NH}, {15, wp2 + OFF_L4, NOUT, NH},
    };
    for (int i = 0; i < 8; ++i) {
        int elems = packs[i].N * packs[i].K;
        nsf_pack_w<<<(elems + 255) / 256, 256, 0, stream>>>(
            (const float*)d_in[packs[i].src], packs[i].dst, packs[i].N, packs[i].K);
    }

    const int blocks = nrows / ROWS;
    // half 1: cond = MLP1(lower = x[:, :32]); transform upper -> out[:, 32:64]; ld1 -> ws
    nsf_half_kernel<<<blocks, 256, LDS_BYTES, stream>>>(
        x, 0, x, 32, wp1,
        (const float*)d_in[2], (const float*)d_in[4],
        (const float*)d_in[6], (const float*)d_in[8],
        yout, 32, ld1, nullptr);
    // half 2: cond = MLP2(upper' = out[:, 32:64]); transform lower -> out[:, 0:32]; logdet = ld1+ld2
    nsf_half_kernel<<<blocks, 256, LDS_BYTES, stream>>>(
        yout, 32, x, 0, wp2,
        (const float*)d_in[10], (const float*)d_in[12],
        (const float*)d_in[14], (const float*)d_in[16],
        yout, 0, ld1, logdet);
}
